// Hyper_DGCNN_72353019068519
// MI455X (gfx1250) — compile-verified, matrix operand order checked by reading, not run on hardware
//
#include <hip/hip_runtime.h>
#include <hip/hip_bf16.h>
#include <math.h>

typedef __attribute__((ext_vector_type(16))) _Float16 v16h;
typedef __attribute__((ext_vector_type(8)))  _Float16 v8h;
typedef __attribute__((ext_vector_type(8)))  float    v8f;

#define HB 8
#define HN 1024
#define HK 20
#define BNP   (HB*HN)        // 8192 points
#define NSAMP (BNP*HK)       // 163840 edge samples
#define FSTRIDE 512
#define CURV 0.01f
#define SQC  0.1f
#define NEGBIG (-3.402823466e38f)

// ---------------------------------------------------------------- utilities
__device__ __forceinline__ void atomic_max_f32(float* addr, float v) {
  if (v >= 0.f) atomicMax((int*)addr, __float_as_int(v));
  else          atomicMin((unsigned int*)addr, __float_as_uint(v));
}

__device__ __forceinline__ v8h cvt8(float4 a, float4 b) {
  v8h r;
  r[0]=(_Float16)a.x; r[1]=(_Float16)a.y; r[2]=(_Float16)a.z; r[3]=(_Float16)a.w;
  r[4]=(_Float16)b.x; r[5]=(_Float16)b.y; r[6]=(_Float16)b.z; r[7]=(_Float16)b.w;
  return r;
}
__device__ __forceinline__ v8h cvt8d(float4 na, float4 nb, float4 ca, float4 cb) {
  v8h r;
  r[0]=(_Float16)(na.x-ca.x); r[1]=(_Float16)(na.y-ca.y);
  r[2]=(_Float16)(na.z-ca.z); r[3]=(_Float16)(na.w-ca.w);
  r[4]=(_Float16)(nb.x-cb.x); r[5]=(_Float16)(nb.y-cb.y);
  r[6]=(_Float16)(nb.z-cb.z); r[7]=(_Float16)(nb.w-cb.w);
  return r;
}
__device__ __forceinline__ v16h combine8(v8h lo, v8h hi8) {
  return __builtin_shufflevector(lo, hi8, 0,1,2,3,4,5,6,7,8,9,10,11,12,13,14,15);
}

// ---------------------------------------------------------------- init
#define STATS_N 6018
__global__ void init_kernel(float* __restrict__ feat, float* __restrict__ outp,
                            float* __restrict__ stats) {
  const int nf = BNP * 448;            // feat columns 64..512 -> -inf (atomic-max targets)
  const int total = nf + BNP + STATS_N;
  for (int i = blockIdx.x*blockDim.x + threadIdx.x; i < total; i += gridDim.x*blockDim.x) {
    if (i < nf) {
      int r = i / 448, c = i % 448;
      feat[(size_t)r * FSTRIDE + 64 + c] = NEGBIG;
    } else if (i < nf + BNP) {
      outp[i - nf] = NEGBIG;
    } else {
      stats[i - nf - BNP] = 0.f;
    }
  }
}

__global__ void cvt_w_kernel(const float* __restrict__ w, _Float16* __restrict__ wh, int n) {
  for (int i = blockIdx.x*blockDim.x + threadIdx.x; i < n; i += gridDim.x*blockDim.x)
    wh[i] = (_Float16)w[i];
}

// ---------------------------------------------------------------- prep: transpose + expmap0
__global__ void prep_kernel(const float* __restrict__ x, float* __restrict__ xt,
                            float* __restrict__ xp) {
  int t = blockIdx.x*blockDim.x + threadIdx.x;
  if (t >= BNP) return;
  int b = t / HN, n = t % HN;
  float v[3], s = 0.f;
#pragma unroll
  for (int c = 0; c < 3; ++c) { v[c] = x[((size_t)b*3 + c)*HN + n]; s += v[c]*v[c]; }
  float nrm = sqrtf(fmaxf(s, 1e-15f));
  float sc  = tanhf(SQC*nrm) / (SQC*nrm);
#pragma unroll
  for (int c = 0; c < 3; ++c) { xt[t*3+c] = v[c]; xp[t*3+c] = sc*v[c]; }
}

// ---------------------------------------------------------------- knn kernels
__global__ void knn3_kernel(const float* __restrict__ pts, int* __restrict__ outIdx) {
  int t = blockIdx.x*blockDim.x + threadIdx.x;
  if (t >= BNP) return;
  int b = t / HN;
  float p0 = pts[t*3+0], p1 = pts[t*3+1], p2 = pts[t*3+2];
  const float* base = pts + (size_t)b * HN * 3;
  float bestd[HK]; int besti[HK];
#pragma unroll
  for (int i = 0; i < HK; ++i) { bestd[i] = NEGBIG; besti[i] = b*HN; }
  for (int m = 0; m < HN; ++m) {
    float d0 = p0 - base[m*3+0], d1 = p1 - base[m*3+1], d2 = p2 - base[m*3+2];
    float pd = -(d0*d0 + d1*d1 + d2*d2);
    if (pd > bestd[HK-1]) {
      int i = HK-1;
      while (i > 0 && bestd[i-1] < pd) { bestd[i] = bestd[i-1]; besti[i] = besti[i-1]; --i; }
      bestd[i] = pd; besti[i] = b*HN + m;
    }
  }
  for (int k = 0; k < HK; ++k) outIdx[(size_t)t*HK + k] = besti[k];
}

template<int C4>
__global__ void knn_vec_kernel(const float* __restrict__ pts, int* __restrict__ outIdx) {
  int t = blockIdx.x*blockDim.x + threadIdx.x;
  if (t >= BNP) return;
  int b = t / HN;
  const float4* pn = (const float4*)(pts + (size_t)t * FSTRIDE);
  float4 pr[C4];
#pragma unroll
  for (int i = 0; i < C4; ++i) pr[i] = pn[i];
  const float* base = pts + (size_t)b * HN * FSTRIDE;
  float bestd[HK]; int besti[HK];
#pragma unroll
  for (int i = 0; i < HK; ++i) { bestd[i] = NEGBIG; besti[i] = b*HN; }
  for (int m = 0; m < HN; ++m) {
    const float4* pm = (const float4*)(base + (size_t)m * FSTRIDE);
    float s = 0.f;
#pragma unroll
    for (int i = 0; i < C4; ++i) {
      float4 q = pm[i];
      float d0 = pr[i].x - q.x, d1 = pr[i].y - q.y, d2 = pr[i].z - q.z, d3 = pr[i].w - q.w;
      s += d0*d0 + d1*d1 + d2*d2 + d3*d3;
    }
    float pd = -s;
    if (pd > bestd[HK-1]) {
      int i = HK-1;
      while (i > 0 && bestd[i-1] < pd) { bestd[i] = bestd[i-1]; besti[i] = besti[i-1]; --i; }
      bestd[i] = pd; besti[i] = b*HN + m;
    }
  }
  for (int k = 0; k < HK; ++k) outIdx[(size_t)t*HK + k] = besti[k];
}

// ---------------------------------------------------------------- stage 1 (hyperbolic, VALU)
__device__ __forceinline__ void s1_compute_u(const float* __restrict__ xp,
                                             const float* __restrict__ W1,
                                             int bn, int nbr, float* u) {
  float xb[3], yb[3];
#pragma unroll
  for (int c = 0; c < 3; ++c) { xb[c] = xp[nbr*3+c]; yb[c] = -xp[bn*3+c]; }
  float x2 = 0.f, y2 = 0.f, xy = 0.f;
#pragma unroll
  for (int c = 0; c < 3; ++c) { x2 += xb[c]*xb[c]; y2 += yb[c]*yb[c]; xy += xb[c]*yb[c]; }
  float ka  = 1.f + 2.f*CURV*xy + CURV*y2;
  float kb  = 1.f - CURV*x2;
  float den = fmaxf(1.f + 2.f*CURV*xy + CURV*CURV*x2*y2, 1e-15f);
  float h6[6];
#pragma unroll
  for (int c = 0; c < 3; ++c) { h6[c] = (ka*xb[c] + kb*yb[c]) / den; h6[3+c] = -yb[c]; }
  float s = 0.f;
#pragma unroll
  for (int c = 0; c < 6; ++c) s += h6[c]*h6[c];
  float nrm = sqrtf(fmaxf(s, 1e-15f));
  float t   = fminf(SQC*nrm, 1.f - 1e-7f);
  float scl = atanhf(t) / (SQC*nrm);
#pragma unroll
  for (int o = 0; o < 64; ++o) {
    float acc = 0.f;
#pragma unroll
    for (int c = 0; c < 6; ++c) acc += W1[o*6+c] * (h6[c]*scl);
    u[o] = acc;
  }
}

__global__ void s1_pass1(const float* __restrict__ xp, const int* __restrict__ idx,
                         const float* __restrict__ W1,
                         float* __restrict__ sum64, float* __restrict__ sq1) {
  int bn = blockIdx.x*blockDim.x + threadIdx.x;
  if (bn >= BNP) return;
  float ls[64]; float lq = 0.f;
#pragma unroll
  for (int o = 0; o < 64; ++o) ls[o] = 0.f;
  for (int k = 0; k < HK; ++k) {
    float u[64];
    s1_compute_u(xp, W1, bn, idx[bn*HK+k], u);
#pragma unroll
    for (int o = 0; o < 64; ++o) { ls[o] += u[o]; lq += u[o]*u[o]; }
  }
#pragma unroll
  for (int o = 0; o < 64; ++o) atomicAdd(&sum64[o], ls[o]);
  atomicAdd(sq1, lq);
}

__global__ void s1_fin(const float* __restrict__ sum64, const float* __restrict__ sq1,
                       float* __restrict__ mu64, float* __restrict__ istd1) {
  if (threadIdx.x || blockIdx.x) return;
  const float cnt = (float)NSAMP;
  float msq = 0.f;
  for (int o = 0; o < 64; ++o) { float m = sum64[o]/cnt; mu64[o] = m; msq += m*m; }
  float var = (sq1[0] - cnt*msq) / cnt;
  istd1[0] = rsqrtf(var + 1e-5f);
}

__global__ void s1_pass2(const float* __restrict__ xp, const int* __restrict__ idx,
                         const float* __restrict__ W1, const float* __restrict__ mu64,
                         const float* __restrict__ istd1, float* __restrict__ feat) {
  int bn = blockIdx.x*blockDim.x + threadIdx.x;
  if (bn >= BNP) return;
  const float is = istd1[0];
  float hmax[64];
#pragma unroll
  for (int o = 0; o < 64; ++o) hmax[o] = NEGBIG;
  for (int k = 0; k < HK; ++k) {
    float u[64];
    s1_compute_u(xp, W1, bn, idx[bn*HK+k], u);
    float s = 0.f;
#pragma unroll
    for (int o = 0; o < 64; ++o) {
      float v = (u[o] - mu64[o]) * is;
      v = v > 0.f ? v : 0.2f*v;                 // tangent-space leaky relu
      u[o] = v; s += v*v;
    }
    float nrm = sqrtf(fmaxf(s, 1e-15f));
    float sc  = tanhf(SQC*nrm) / (SQC*nrm);     // expmap0
#pragma unroll
    for (int o = 0; o < 64; ++o) hmax[o] = fmaxf(hmax[o], sc*u[o]);
  }
  float s = 0.f;
#pragma unroll
  for (int o = 0; o < 64; ++o) s += hmax[o]*hmax[o];
  float nrm = sqrtf(fmaxf(s, 1e-15f));
  float t   = fminf(SQC*nrm, 1.f - 1e-7f);
  float sc  = atanhf(t) / (SQC*nrm);            // logmap0
#pragma unroll
  for (int o = 0; o < 64; ++o) feat[(size_t)bn*FSTRIDE + o] = sc*hmax[o];
}

// ---------------------------------------------------------------- WMMA GEMM core
// One wave computes a 16-row x 64-channel tile (4 accumulators).
// All loads of a k-step are grouped (one clause), B fragments live in 4 distinct
// register sets, then 4 WMMAs issue back-to-back (no WAR nop chains).
// C>0: edge mode; K-reduction split into [nbr-ctr | ctr] halves. C==0: dense rows.
template<int C, int TWOC>
__device__ __forceinline__ void compute_acc(const float* __restrict__ feat, int inOff,
                                            const int* __restrict__ idx,
                                            const _Float16* __restrict__ wh,
                                            int och0, int lane, v8f acc[4]) {
  const int m = lane & 15, hi = lane >> 4;
  const int s = blockIdx.x * 16 + m;
  const int bnRow = (C > 0) ? (s / HK) : s;
  const float* fc = feat + (size_t)bnRow * FSTRIDE + inOff;
  const _Float16* wr[4];
  v8f z = {};
#pragma unroll
  for (int t = 0; t < 4; ++t) {
    wr[t] = wh + (size_t)(och0 + t*16 + m) * TWOC;
    acc[t] = z;
  }
  if (C > 0) {
    const float* fn = feat + (size_t)idx[s] * FSTRIDE + inOff;
#pragma unroll 2
    for (int k0 = 0; k0 < C; k0 += 32) {        // first half: nbr - ctr
      const int c0 = k0 + hi*8;
      v16h bvv[4];
#pragma unroll
      for (int t = 0; t < 4; ++t)
        bvv[t] = combine8(*(const v8h*)(wr[t]+c0), *(const v8h*)(wr[t]+c0+16));
      float4 n0 = *(const float4*)(fn+c0),    n1 = *(const float4*)(fn+c0+4);
      float4 e0 = *(const float4*)(fc+c0),    e1 = *(const float4*)(fc+c0+4);
      float4 n2 = *(const float4*)(fn+c0+16), n3 = *(const float4*)(fn+c0+20);
      float4 e2 = *(const float4*)(fc+c0+16), e3 = *(const float4*)(fc+c0+20);
      v16h av = combine8(cvt8d(n0,n1,e0,e1), cvt8d(n2,n3,e2,e3));
#pragma unroll
      for (int t = 0; t < 4; ++t)
        acc[t] = __builtin_amdgcn_wmma_f32_16x16x32_f16(false, av, false, bvv[t],
                                                        (short)0, acc[t], false, false);
    }
#pragma unroll 2
    for (int k0 = 0; k0 < C; k0 += 32) {        // second half: ctr, weight cols C..2C
      const int c0 = k0 + hi*8;
      v16h bvv[4];
#pragma unroll
      for (int t = 0; t < 4; ++t)
        bvv[t] = combine8(*(const v8h*)(wr[t]+C+c0), *(const v8h*)(wr[t]+C+c0+16));
      float4 e0 = *(const float4*)(fc+c0),    e1 = *(const float4*)(fc+c0+4);
      float4 e2 = *(const float4*)(fc+c0+16), e3 = *(const float4*)(fc+c0+20);
      v16h av = combine8(cvt8(e0,e1), cvt8(e2,e3));
#pragma unroll
      for (int t = 0; t < 4; ++t)
        acc[t] = __builtin_amdgcn_wmma_f32_16x16x32_f16(false, av, false, bvv[t],
                                                        (short)0, acc[t], false, false);
    }
  } else {
#pragma unroll 2
    for (int k0 = 0; k0 < TWOC; k0 += 32) {     // dense row
      const int c0 = k0 + hi*8;
      v16h bvv[4];
#pragma unroll
      for (int t = 0; t < 4; ++t)
        bvv[t] = combine8(*(const v8h*)(wr[t]+c0), *(const v8h*)(wr[t]+c0+16));
      float4 e0 = *(const float4*)(fc+c0),    e1 = *(const float4*)(fc+c0+4);
      float4 e2 = *(const float4*)(fc+c0+16), e3 = *(const float4*)(fc+c0+20);
      v16h av = combine8(cvt8(e0,e1), cvt8(e2,e3));
#pragma unroll
      for (int t = 0; t < 4; ++t)
        acc[t] = __builtin_amdgcn_wmma_f32_16x16x32_f16(false, av, false, bvv[t],
                                                        (short)0, acc[t], false, false);
    }
  }
}

template<int C, int TWOC>
__global__ void gemm_pass1_k(const float* __restrict__ feat, int inOff,
                             const int* __restrict__ idx,
                             const _Float16* __restrict__ wh,
                             float* __restrict__ sumv, float* __restrict__ sqv) {
  const int lane = threadIdx.x & 31;
  const int och0 = blockIdx.y * 64;
  v8f acc[4];
  compute_acc<C, TWOC>(feat, inOff, idx, wh, och0, lane, acc);
#pragma unroll
  for (int t = 0; t < 4; ++t) {
    int o = och0 + t*16 + (lane & 15);
    float ls = 0.f, lq = 0.f;
#pragma unroll
    for (int r = 0; r < 8; ++r) { float y = acc[t][r]; ls += y; lq += y*y; }
    ls += __shfl_xor(ls, 16, 32);               // fold half-lanes (same channel o)
    lq += __shfl_xor(lq, 16, 32);
    if (lane < 16) {
      atomicAdd(&sumv[o], ls);
      atomicAdd(&sqv[o], lq);
    }
  }
}

template<int C, int TWOC>
__global__ void gemm_pass2_k(const float* __restrict__ feat, int inOff,
                             const int* __restrict__ idx,
                             const _Float16* __restrict__ wh,
                             const float* __restrict__ mu, const float* __restrict__ istd,
                             const float* __restrict__ gg, const float* __restrict__ bb,
                             float* __restrict__ outp, int outStride, int outOff, int outDiv) {
  const int lane = threadIdx.x & 31;
  const int hi = lane >> 4;
  const int och0 = blockIdx.y * 64;
  v8f acc[4];
  compute_acc<C, TWOC>(feat, inOff, idx, wh, och0, lane, acc);
#pragma unroll
  for (int t = 0; t < 4; ++t) {
    int o = och0 + t*16 + (lane & 15);
    const float muo = mu[o], iso = istd[o], go = gg[o], bo = bb[o];
#pragma unroll
    for (int r = 0; r < 8; ++r) {
      int row = blockIdx.x * 16 + r + hi*8;     // D layout: VGPR r holds M=r (+8 hi lanes)
      int grp = row / outDiv;
      float y = (acc[t][r] - muo) * iso * go + bo;
      y = y > 0.f ? y : 0.2f*y;
      atomic_max_f32(&outp[(size_t)grp * outStride + outOff + o], y);
    }
  }
}

__global__ void fin_stats(const float* __restrict__ sumv, const float* __restrict__ sqv,
                          float* __restrict__ mu, float* __restrict__ istd,
                          int Cout, float cnt) {
  int o = blockIdx.x*blockDim.x + threadIdx.x;
  if (o >= Cout) return;
  float m = sumv[o]/cnt;
  float v = sqv[o]/cnt - m*m;
  mu[o] = m;
  istd[o] = rsqrtf(v + 1e-5f);
}

// ---------------------------------------------------------------- launch
extern "C" void kernel_launch(void* const* d_in, const int* in_sizes, int n_in,
                              void* d_out, int out_size, void* d_ws, size_t ws_size,
                              hipStream_t stream) {
  const float* x  = (const float*)d_in[0];
  const float* W1 = (const float*)d_in[1];
  const float* w2 = (const float*)d_in[2];
  const float* g2 = (const float*)d_in[3];
  const float* b2 = (const float*)d_in[4];
  const float* w3 = (const float*)d_in[5];
  const float* g3 = (const float*)d_in[6];
  const float* b3 = (const float*)d_in[7];
  const float* w4 = (const float*)d_in[8];
  const float* g4 = (const float*)d_in[9];
  const float* b4 = (const float*)d_in[10];
  const float* w5 = (const float*)d_in[11];
  const float* g5 = (const float*)d_in[12];
  const float* b5 = (const float*)d_in[13];
  float* out = (float*)d_out;

  float* ws   = (float*)d_ws;
  float* FEAT = ws;                         // (BNP, 512)
  float* XP   = FEAT + (size_t)BNP*FSTRIDE; // (BNP, 3)
  float* XT   = XP + BNP*3;                 // (BNP, 3)
  int* IDX1 = (int*)(XT + BNP*3);
  int* IDX2 = IDX1 + NSAMP;
  int* IDX3 = IDX2 + NSAMP;
  int* IDX4 = IDX3 + NSAMP;
  float* STATS = (float*)(IDX4 + NSAMP);
  float* p = STATS;
  float* S1_SUM = p; p += 64;  float* S1_SQ = p; p += 1;
  float* S1_MU  = p; p += 64;  float* S1_IS = p; p += 1;
  float* S2_SUM = p; p += 64;  float* S2_SQ = p; p += 64;
  float* S2_MU  = p; p += 64;  float* S2_IS = p; p += 64;
  float* S3_SUM = p; p += 128; float* S3_SQ = p; p += 128;
  float* S3_MU  = p; p += 128; float* S3_IS = p; p += 128;
  float* S4_SUM = p; p += 256; float* S4_SQ = p; p += 256;
  float* S4_MU  = p; p += 256; float* S4_IS = p; p += 256;
  float* S5_SUM = p; p += 1024; float* S5_SQ = p; p += 1024;
  float* S5_MU  = p; p += 1024; float* S5_IS = p; p += 1024;
  _Float16* WH2 = (_Float16*)(STATS + 6144);   // f16 weights (L2-resident)
  _Float16* WH3 = WH2 + 64*128;
  _Float16* WH4 = WH3 + 128*128;
  _Float16* WH5 = WH4 + 256*256;

  const dim3 wblk(32);
  const float cntE = (float)NSAMP, cntF = (float)BNP;

  // init + weight conversion + prep + knn on raw coords
  init_kernel<<<2048, 256, 0, stream>>>(FEAT, out, STATS);
  cvt_w_kernel<<<32, 256, 0, stream>>>(w2, WH2, 64*128);
  cvt_w_kernel<<<64, 256, 0, stream>>>(w3, WH3, 128*128);
  cvt_w_kernel<<<256, 256, 0, stream>>>(w4, WH4, 256*256);
  cvt_w_kernel<<<1024, 256, 0, stream>>>(w5, WH5, 1024*512);
  prep_kernel<<<(BNP+255)/256, 256, 0, stream>>>(x, XT, XP);
  knn3_kernel<<<(BNP+255)/256, 256, 0, stream>>>(XT, IDX1);

  // stage 1: hyperbolic edge conv -> feat cols [0,64)
  s1_pass1<<<(BNP+127)/128, 128, 0, stream>>>(XP, IDX1, W1, S1_SUM, S1_SQ);
  s1_fin<<<1, 1, 0, stream>>>(S1_SUM, S1_SQ, S1_MU, S1_IS);
  s1_pass2<<<(BNP+127)/128, 128, 0, stream>>>(XP, IDX1, W1, S1_MU, S1_IS, FEAT);

  // stage 2: x1(64) -> x2(64) at cols [64,128)
  knn_vec_kernel<16><<<(BNP+255)/256, 256, 0, stream>>>(FEAT + 0, IDX2);
  {
    dim3 g(NSAMP/16, 1);
    gemm_pass1_k<64,128><<<g, wblk, 0, stream>>>(FEAT, 0, IDX2, WH2, S2_SUM, S2_SQ);
    fin_stats<<<1, 64, 0, stream>>>(S2_SUM, S2_SQ, S2_MU, S2_IS, 64, cntE);
    gemm_pass2_k<64,128><<<g, wblk, 0, stream>>>(FEAT, 0, IDX2, WH2,
                                                 S2_MU, S2_IS, g2, b2, FEAT, FSTRIDE, 64, HK);
  }

  // stage 3: x2(64) -> x3(128) at cols [128,256)
  knn_vec_kernel<16><<<(BNP+255)/256, 256, 0, stream>>>(FEAT + 64, IDX3);
  {
    dim3 g(NSAMP/16, 2);
    gemm_pass1_k<64,128><<<g, wblk, 0, stream>>>(FEAT, 64, IDX3, WH3, S3_SUM, S3_SQ);
    fin_stats<<<2, 64, 0, stream>>>(S3_SUM, S3_SQ, S3_MU, S3_IS, 128, cntE);
    gemm_pass2_k<64,128><<<g, wblk, 0, stream>>>(FEAT, 64, IDX3, WH3,
                                                 S3_MU, S3_IS, g3, b3, FEAT, FSTRIDE, 128, HK);
  }

  // stage 4: x3(128) -> x4(256) at cols [256,512)
  knn_vec_kernel<32><<<(BNP+255)/256, 256, 0, stream>>>(FEAT + 128, IDX4);
  {
    dim3 g(NSAMP/16, 4);
    gemm_pass1_k<128,256><<<g, wblk, 0, stream>>>(FEAT, 128, IDX4, WH4, S4_SUM, S4_SQ);
    fin_stats<<<4, 64, 0, stream>>>(S4_SUM, S4_SQ, S4_MU, S4_IS, 256, cntE);
    gemm_pass2_k<128,256><<<g, wblk, 0, stream>>>(FEAT, 128, IDX4, WH4,
                                                  S4_MU, S4_IS, g4, b4, FEAT, FSTRIDE, 256, HK);
  }

  // stage 5: dense feat(512) -> 1024, BN + lrelu + max over N -> d_out (8,1024)
  {
    dim3 g(BNP/16, 16);
    gemm_pass1_k<0,512><<<g, wblk, 0, stream>>>(FEAT, 0, nullptr, WH5, S5_SUM, S5_SQ);
    fin_stats<<<16, 64, 0, stream>>>(S5_SUM, S5_SQ, S5_MU, S5_IS, 1024, cntF);
    gemm_pass2_k<0,512><<<g, wblk, 0, stream>>>(FEAT, 0, nullptr, WH5,
                                                S5_MU, S5_IS, g5, b5, out, 1024, 0, HN);
  }
}
